// NeRFRenderer_82927228551225
// MI455X (gfx1250) — compile-verified, matrix operand order checked by reading, not run on hardware
//
#include <hip/hip_runtime.h>
#include <hip/hip_bf16.h>
#include <math.h>

typedef __attribute__((ext_vector_type(16))) _Float16 v16h;
typedef __attribute__((ext_vector_type(8)))  _Float16 v8h;
typedef __attribute__((ext_vector_type(8)))  float    v8f;
typedef _Float16 half_t;

#define MIN_NEAR 0.05f
#define NS 128            // num_steps
#define US 128            // upsample_steps
#define TT 256            // NS + US
#define HDIM 256
#define GEO 15
#define PTILES 8          // 32-row M-tiles per workgroup (256 points/WG)

// ---------------------------------------------------------------------------
// WMMA helpers (CDNA5 16x16x32 f16 -> f32)
// ---------------------------------------------------------------------------
__device__ inline v8f wmma16(v16h a, v16h b, v8f c) {
  return __builtin_amdgcn_wmma_f32_16x16x32_f16(false, a, false, b, (short)0, c,
                                                false, false);
}

// A fragment from LDS, row-major [16][stride] f16, K-chunk k0.
// 16-bit A 16x32 layout: lane m (0-15 / 16-31 -> hb), elems 0..7 at k0+hb*8,
// elems 8..15 at k0+16+hb*8 -> two contiguous v8h (ds_load_b128 each).
__device__ inline v16h load_a_lds(const half_t* __restrict__ src, int k0, int stride) {
  const int lane = threadIdx.x & 31;
  const int hb   = lane >> 4;
  const int m    = lane & 15;
  const half_t* p = src + m * stride + k0 + hb * 8;
  v8h lo = *(const v8h*)p;
  v8h hi = *(const v8h*)(p + 16);
  return __builtin_shufflevector(lo, hi, 0, 1, 2, 3, 4, 5, 6, 7,
                                 8, 9, 10, 11, 12, 13, 14, 15);
}

// B fragment from pre-packed weights: lane's 16 halfs are contiguous (32B),
// fragment index ft = kc*ntiles + tn -> two global_load_b128 per lane,
// wave reads 1KB contiguous.
__device__ inline v16h load_b_packed(const half_t* __restrict__ Wp, int kc, int tn,
                                     int ntiles) {
  const int lane = threadIdx.x & 31;
  const half_t* p = Wp + ((((kc * ntiles + tn) << 5) + lane) << 4);
  v8h lo = *(const v8h*)p;
  v8h hi = *(const v8h*)(p + 8);
  return __builtin_shufflevector(lo, hi, 0, 1, 2, 3, 4, 5, 6, 7,
                                 8, 9, 10, 11, 12, 13, 14, 15);
}

// Software-pipelined K-sweep: prefetch next chunk's A fragments before
// consuming the current ones, so ds_load latency overlaps the WMMA pair.
__device__ inline void ksweep_2m(const half_t* __restrict__ hsrc, const v16h* bc,
                                 v8f& a0, v8f& a1) {
  v16h fa0 = load_a_lds(hsrc, 0, HDIM);
  v16h fa1 = load_a_lds(hsrc + 16 * HDIM, 0, HDIM);
#pragma unroll
  for (int kc = 0; kc < 8; ++kc) {
    v16h na0, na1;
    if (kc < 7) {
      na0 = load_a_lds(hsrc, (kc + 1) * 32, HDIM);
      na1 = load_a_lds(hsrc + 16 * HDIM, (kc + 1) * 32, HDIM);
    }
    a0 = wmma16(fa0, bc[kc], a0);
    a1 = wmma16(fa1, bc[kc], a1);
    if (kc < 7) { fa0 = na0; fa1 = na1; }
  }
}

// ---------------------------------------------------------------------------
// Pack f32 weights [K=256][src_cols] into fragment-native f16 layout.
// ntiles N-tiles of 16; zero-pads columns >= src_cols (for cW3: 3 -> 16).
// ---------------------------------------------------------------------------
__global__ void pack_b_kernel(const float* __restrict__ W, half_t* __restrict__ dst,
                              int ntiles, int src_cols, int total) {
  int i = blockIdx.x * blockDim.x + threadIdx.x;
  if (i >= total) return;
  int e    = i & 15;
  int lane = (i >> 4) & 31;
  int ft   = i >> 9;
  int tn   = ft % ntiles;
  int kc   = ft / ntiles;
  int hb   = lane >> 4;
  int k = kc * 32 + ((e < 8) ? (hb * 8 + e) : (16 + hb * 8 + (e - 8)));
  int n = tn * 16 + (lane & 15);
  float v = (n < src_cols) ? W[k * src_cols + n] : 0.0f;
  dst[i] = (half_t)v;
}

// ---------------------------------------------------------------------------
// Ray setup: sphere intersection + coarse linspace samples
// ---------------------------------------------------------------------------
__global__ void ray_setup_kernel(const float* __restrict__ ro,
                                 const float* __restrict__ rd,
                                 float* __restrict__ zc,
                                 float* __restrict__ xyzc, int total) {
  int i = blockIdx.x * blockDim.x + threadIdx.x;
  if (i >= total) return;
  int ray = i >> 7, s = i & (NS - 1);
  float ox = ro[ray * 3 + 0], oy = ro[ray * 3 + 1], oz = ro[ray * 3 + 2];
  float dx = rd[ray * 3 + 0], dy = rd[ray * 3 + 1], dz = rd[ray * 3 + 2];
  float b = ox * dx + oy * dy + oz * dz;
  float c = ox * ox + oy * oy + oz * oz - 1.0f;
  float disc = fmaxf(b * b - c, 0.0f);
  float zmax = fmaxf(-b + sqrtf(disc), MIN_NEAR + 1e-3f);
  float zmin = MIN_NEAR;
  float t = (float)s / (float)(NS - 1);
  float z = zmin + (zmax - zmin) * t;
  z = fminf(fmaxf(z, zmin), zmax);
  zc[i] = z;
  xyzc[i * 3 + 0] = ox + dx * z;
  xyzc[i * 3 + 1] = oy + dy * z;
  xyzc[i * 3 + 2] = oz + dz * z;
}

// ---------------------------------------------------------------------------
// Density MLP: 3 -> 256 -> 256 -> 16 (sigma + 15 geo)
// 512 threads = 16 waves; each wave owns one N-tile and register-caches its
// whole K-sweep of W2 fragments; 8 M-tiles of 32 points per workgroup.
// ---------------------------------------------------------------------------
__global__ __launch_bounds__(512) void density_mlp_kernel(
    const float* __restrict__ xyz,      // [M][3]
    const float* __restrict__ W1,       // [3][256] f32
    const float* __restrict__ b1,       // [256]
    const half_t* __restrict__ W2p,     // packed [8][16][32][16] f16
    const float* __restrict__ b2,       // [256]
    const half_t* __restrict__ W3p,     // packed [8][1][32][16] f16
    const float* __restrict__ b3,       // [16]
    float* __restrict__ sigma,          // [M]
    half_t* __restrict__ geo) {         // [M][15] f16
  __shared__ __align__(16) half_t h1[32 * HDIM];
  __shared__ __align__(16) half_t h2[32 * HDIM];
  __shared__ float xloc[32][3];

  const int tid       = threadIdx.x;
  const int wave      = tid >> 5;
  const int blockBase = blockIdx.x * (32 * PTILES);

  // register-cache this wave's B fragments for all 8 K-chunks (64 VGPRs)
  v16h bc[8];
#pragma unroll
  for (int kc = 0; kc < 8; ++kc) bc[kc] = load_b_packed(W2p, kc, wave, 16);

  for (int p = 0; p < PTILES; ++p) {
    const int base = blockBase + p * 32;

    if (tid < 96) {
      int m = tid / 3, k = tid % 3;
      xloc[m][k] = xyz[(base + m) * 3 + k];
    }
    __syncthreads();

    // ---- layer 1 (K=3, scalar) -> relu -> packed f16 LDS stores
    {
      int m = tid >> 4, n0 = (tid & 15) * 16;
      float x0 = xloc[m][0], x1 = xloc[m][1], x2 = xloc[m][2];
      v8h s0, s1;
#pragma unroll
      for (int e = 0; e < 8; ++e) {
        int n = n0 + e;
        float h = x0 * W1[n] + x1 * W1[HDIM + n] + x2 * W1[2 * HDIM + n] + b1[n];
        s0[e] = (half_t)fmaxf(h, 0.0f);
      }
#pragma unroll
      for (int e = 0; e < 8; ++e) {
        int n = n0 + 8 + e;
        float h = x0 * W1[n] + x1 * W1[HDIM + n] + x2 * W1[2 * HDIM + n] + b1[n];
        s1[e] = (half_t)fmaxf(h, 0.0f);
      }
      *(v8h*)(h1 + m * HDIM + n0)     = s0;
      *(v8h*)(h1 + m * HDIM + n0 + 8) = s1;
    }
    __syncthreads();

    // ---- layer 2 (256x256) via WMMA with cached B, pipelined A loads
    {
      v8f a0 = {}, a1 = {};
      ksweep_2m(h1, bc, a0, a1);
      const int lane = tid & 31, hb = lane >> 4;
      const int n = (wave << 4) + (lane & 15);
#pragma unroll
      for (int r = 0; r < 8; ++r) {
        int m = r + 8 * hb;
        h2[m * HDIM + n]        = (half_t)fmaxf(a0[r] + b2[n], 0.0f);
        h2[(m + 16) * HDIM + n] = (half_t)fmaxf(a1[r] + b2[n], 0.0f);
      }
    }
    __syncthreads();

    // ---- layer 3 (256 -> 16): waves 0 and 1, one M-subtile each
    if (tid < 64) {
      const int mb = tid >> 5;
      const half_t* hsrc = h2 + mb * 16 * HDIM;
      v8f acc = {};
      v16h fa = load_a_lds(hsrc, 0, HDIM);
      v16h fb = load_b_packed(W3p, 0, 0, 1);
#pragma unroll
      for (int kc = 0; kc < 8; ++kc) {
        v16h na, nb;
        if (kc < 7) {
          na = load_a_lds(hsrc, (kc + 1) * 32, HDIM);
          nb = load_b_packed(W3p, kc + 1, 0, 1);
        }
        acc = wmma16(fa, fb, acc);
        if (kc < 7) { fa = na; fb = nb; }
      }
      const int lane = tid & 31, hb = lane >> 4, n = lane & 15;
#pragma unroll
      for (int r = 0; r < 8; ++r) {
        int row = base + mb * 16 + r + 8 * hb;
        float v = acc[r] + b3[n];
        if (n == 0) {
          sigma[row] = expf(fminf(fmaxf(v, -15.0f), 15.0f));  // trunc_exp
        } else {
          geo[row * GEO + (n - 1)] = (half_t)v;
        }
      }
    }
    __syncthreads();
  }
}

// ---------------------------------------------------------------------------
// Upsample: coarse weights -> deterministic inverse-CDF resample (per ray)
// ---------------------------------------------------------------------------
__global__ void upsample_kernel(const float* __restrict__ ro,
                                const float* __restrict__ rd,
                                const float* __restrict__ zc,
                                const float* __restrict__ sigc,
                                float* __restrict__ zf,
                                float* __restrict__ xyzf, int N) {
  int ray = blockIdx.x * blockDim.x + threadIdx.x;
  if (ray >= N) return;

  float zarr[NS];
  float sg[NS];
  for (int i = 0; i < NS; ++i) zarr[i] = zc[ray * NS + i];
  for (int i = 0; i < NS; ++i) sg[i]   = sigc[ray * NS + i];

  float w[NS];
  float T = 1.0f;
  for (int i = 0; i < NS; ++i) {
    float d = (i < NS - 1) ? (zarr[i + 1] - zarr[i]) : (zarr[NS - 1] - zarr[NS - 2]);
    float alpha = 1.0f - expf(-d * sg[i]);
    w[i] = alpha * T;
    T *= (1.0f - alpha + 1e-15f);
  }

  float cdf[NS - 1];
  float s = 0.0f;
  for (int j = 1; j <= NS - 2; ++j) s += w[j] + 1e-5f;
  cdf[0] = 0.0f;
  float acc = 0.0f;
  for (int j = 1; j <= NS - 2; ++j) {
    acc += (w[j] + 1e-5f) / s;
    cdf[j] = acc;
  }

  const float ox = ro[ray * 3 + 0], oy = ro[ray * 3 + 1], oz = ro[ray * 3 + 2];
  const float dx = rd[ray * 3 + 0], dy = rd[ray * 3 + 1], dz = rd[ray * 3 + 2];

  int idx = 0;
  const float u0 = 0.5f / (float)US;
  const float du = (1.0f - 1.0f / (float)US) / (float)(US - 1);
  for (int k = 0; k < US; ++k) {
    float u = u0 + du * (float)k;
    while (idx < NS - 1 && cdf[idx] <= u) idx++;
    int below = idx - 1; if (below < 0) below = 0; if (below > NS - 2) below = NS - 2;
    int above = idx;                               if (above > NS - 2) above = NS - 2;
    float c0 = cdf[below], c1 = cdf[above];
    float bb0 = 0.5f * (zarr[below] + zarr[below + 1]);
    float bb1 = 0.5f * (zarr[above] + zarr[above + 1]);
    float denom = (c1 - c0 < 1e-5f) ? 1.0f : (c1 - c0);
    float tt = (u - c0) / denom;
    float nz = bb0 + tt * (bb1 - bb0);
    zf[ray * US + k] = nz;
    xyzf[(ray * US + k) * 3 + 0] = ox + dx * nz;
    xyzf[(ray * US + k) * 3 + 1] = oy + dy * nz;
    xyzf[(ray * US + k) * 3 + 2] = oz + dz * nz;
  }
}

// ---------------------------------------------------------------------------
// Merge (stable two-way, ties -> coarse) + transmittance weights (per ray)
// ---------------------------------------------------------------------------
__global__ void merge_kernel(const float* __restrict__ zc,
                             const float* __restrict__ zf,
                             const float* __restrict__ sigc,
                             const float* __restrict__ sigf,
                             float* __restrict__ zs,
                             int* __restrict__ order,
                             float* __restrict__ wts, int N) {
  int ray = blockIdx.x * blockDim.x + threadIdx.x;
  if (ray >= N) return;
  int i = 0, j = 0;
  for (int t = 0; t < TT; ++t) {
    float a = (i < NS) ? zc[ray * NS + i] : 3.4e38f;
    float b = (j < US) ? zf[ray * US + j] : 3.4e38f;
    if (a <= b) { zs[ray * TT + t] = a; order[ray * TT + t] = i; ++i; }
    else        { zs[ray * TT + t] = b; order[ray * TT + t] = NS + j; ++j; }
  }
  float T = 1.0f, prevd = 0.0f;
  for (int t = 0; t < TT; ++t) {
    float d;
    if (t < TT - 1) { d = zs[ray * TT + t + 1] - zs[ray * TT + t]; prevd = d; }
    else            { d = prevd; }
    int o = order[ray * TT + t];
    float sgv = (o < NS) ? sigc[ray * NS + o] : sigf[ray * US + (o - NS)];
    float alpha = 1.0f - expf(-d * sgv);
    wts[ray * TT + t] = alpha * T;
    T *= (1.0f - alpha + 1e-15f);
  }
}

// ---------------------------------------------------------------------------
// Color MLP: 18 -> 256 -> 256 -> 3 (sigmoid); same blocking as density MLP
// ---------------------------------------------------------------------------
__global__ __launch_bounds__(512) void color_mlp_kernel(
    const float* __restrict__ rd,       // [N][3]
    const int* __restrict__ order,      // [N*256]
    const half_t* __restrict__ geoc,    // [N*128][15]
    const half_t* __restrict__ geof,    // [N*128][15]
    const float* __restrict__ W1,       // [18][256] f32
    const float* __restrict__ b1,       // [256]
    const half_t* __restrict__ W2p,     // packed
    const float* __restrict__ b2,       // [256]
    const half_t* __restrict__ W3p,     // packed (padded cW3)
    const float* __restrict__ b3,       // [3]
    float* __restrict__ rgb) {          // [M][3]
  __shared__ __align__(16) half_t h1[32 * HDIM];
  __shared__ __align__(16) half_t h2[32 * HDIM];
  __shared__ float feat[32][18];

  const int tid       = threadIdx.x;
  const int wave      = tid >> 5;
  const int blockBase = blockIdx.x * (32 * PTILES);

  v16h bc[8];
#pragma unroll
  for (int kc = 0; kc < 8; ++kc) bc[kc] = load_b_packed(W2p, kc, wave, 16);

  for (int p = 0; p < PTILES; ++p) {
    const int base = blockBase + p * 32;

    if (tid < 32) {
      int pt  = base + tid;
      int ray = pt >> 8;
      int o   = order[pt];
      const half_t* g = (o < NS) ? &geoc[(ray * NS + o) * GEO]
                                 : &geof[(ray * US + (o - NS)) * GEO];
      feat[tid][0] = rd[ray * 3 + 0];
      feat[tid][1] = rd[ray * 3 + 1];
      feat[tid][2] = rd[ray * 3 + 2];
#pragma unroll
      for (int k = 0; k < GEO; ++k) feat[tid][3 + k] = (float)g[k];
    }
    __syncthreads();

    // ---- layer 1 (K=18, scalar)
    {
      int m = tid >> 4, n0 = (tid & 15) * 16;
      v8h s0, s1;
#pragma unroll
      for (int e = 0; e < 8; ++e) {
        int n = n0 + e;
        float h = b1[n];
#pragma unroll
        for (int k = 0; k < 18; ++k) h += feat[m][k] * W1[k * HDIM + n];
        s0[e] = (half_t)fmaxf(h, 0.0f);
      }
#pragma unroll
      for (int e = 0; e < 8; ++e) {
        int n = n0 + 8 + e;
        float h = b1[n];
#pragma unroll
        for (int k = 0; k < 18; ++k) h += feat[m][k] * W1[k * HDIM + n];
        s1[e] = (half_t)fmaxf(h, 0.0f);
      }
      *(v8h*)(h1 + m * HDIM + n0)     = s0;
      *(v8h*)(h1 + m * HDIM + n0 + 8) = s1;
    }
    __syncthreads();

    // ---- layer 2 via WMMA with cached B, pipelined A loads
    {
      v8f a0 = {}, a1 = {};
      ksweep_2m(h1, bc, a0, a1);
      const int lane = tid & 31, hb = lane >> 4;
      const int n = (wave << 4) + (lane & 15);
#pragma unroll
      for (int r = 0; r < 8; ++r) {
        int m = r + 8 * hb;
        h2[m * HDIM + n]        = (half_t)fmaxf(a0[r] + b2[n], 0.0f);
        h2[(m + 16) * HDIM + n] = (half_t)fmaxf(a1[r] + b2[n], 0.0f);
      }
    }
    __syncthreads();

    // ---- layer 3 (256 -> 16-padded, first 3 used) + sigmoid
    if (tid < 64) {
      const int mb = tid >> 5;
      const half_t* hsrc = h2 + mb * 16 * HDIM;
      v8f acc = {};
      v16h fa = load_a_lds(hsrc, 0, HDIM);
      v16h fb = load_b_packed(W3p, 0, 0, 1);
#pragma unroll
      for (int kc = 0; kc < 8; ++kc) {
        v16h na, nb;
        if (kc < 7) {
          na = load_a_lds(hsrc, (kc + 1) * 32, HDIM);
          nb = load_b_packed(W3p, kc + 1, 0, 1);
        }
        acc = wmma16(fa, fb, acc);
        if (kc < 7) { fa = na; fb = nb; }
      }
      const int lane = tid & 31, hb = lane >> 4, n = lane & 15;
      if (n < 3) {
#pragma unroll
        for (int r = 0; r < 8; ++r) {
          int row = base + mb * 16 + r + 8 * hb;
          float v = acc[r] + b3[n];
          rgb[row * 3 + n] = 1.0f / (1.0f + expf(-v));
        }
      }
    }
    __syncthreads();
  }
}

// ---------------------------------------------------------------------------
// Composite + distortion loss (one workgroup per ray)
// ---------------------------------------------------------------------------
__device__ float block_reduce(float v, float* red) {
  int t = threadIdx.x;
  red[t] = v;
  __syncthreads();
  for (int s = 128; s > 0; s >>= 1) {
    if (t < s) red[t] += red[t + s];
    __syncthreads();
  }
  float r = red[0];
  __syncthreads();
  return r;
}

__global__ __launch_bounds__(256) void composite_kernel(
    const float* __restrict__ zs, const float* __restrict__ wts,
    const float* __restrict__ rgb, float* __restrict__ out) {
  __shared__ float sw[TT];
  __shared__ float sm[TT];
  __shared__ float red[TT];

  const int ray = blockIdx.x;
  const int t   = threadIdx.x;
  const int p   = ray * TT + t;

  float w = wts[p];
  float z = zs[p];
  bool mask = w > 1e-4f;
  float r = mask ? rgb[p * 3 + 0] : 0.0f;
  float g = mask ? rgb[p * 3 + 1] : 0.0f;
  float b = mask ? rgb[p * 3 + 2] : 0.0f;

  const float sample_dist = 1.0f / (float)NS;
  float znext = (t < TT - 1) ? zs[p + 1] : sample_dist;
  sw[t] = w;
  sm[t] = 0.5f * z + 0.5f * znext;
  __syncthreads();

  float wsum  = block_reduce(w, red);
  float depth = block_reduce(w * z, red);
  float ir    = block_reduce(w * r, red);
  float ig    = block_reduce(w * g, red);
  float ib    = block_reduce(w * b, red);

  float mt = sm[t];
  float s = 0.0f;
  for (int j = 0; j < TT; ++j) s += fabsf(mt - sm[j]) * sw[j];
  float pl = block_reduce(w * s, red);

  if (t == 0) {
    const float bg = 206.0f / 255.0f;
    float loss = pl / (depth + 1e-6f);
    if (loss < 1e-3f) loss = 0.0f;
    out[ray * 5 + 0] = ir + (1.0f - wsum) * bg;
    out[ray * 5 + 1] = ig + (1.0f - wsum) * bg;
    out[ray * 5 + 2] = ib + (1.0f - wsum) * bg;
    out[ray * 5 + 3] = depth;
    out[ray * 5 + 4] = loss;
  }
}

// ---------------------------------------------------------------------------
// Launch
// ---------------------------------------------------------------------------
extern "C" void kernel_launch(void* const* d_in, const int* in_sizes, int n_in,
                              void* d_out, int out_size, void* d_ws, size_t ws_size,
                              hipStream_t stream) {
  const float* rays_o = (const float*)d_in[0];
  const float* rays_d = (const float*)d_in[1];
  const float* dW1 = (const float*)d_in[2];
  const float* db1 = (const float*)d_in[3];
  const float* dW2 = (const float*)d_in[4];
  const float* db2 = (const float*)d_in[5];
  const float* dW3 = (const float*)d_in[6];
  const float* db3 = (const float*)d_in[7];
  const float* cW1 = (const float*)d_in[8];
  const float* cb1 = (const float*)d_in[9];
  const float* cW2 = (const float*)d_in[10];
  const float* cb2 = (const float*)d_in[11];
  const float* cW3 = (const float*)d_in[12];
  const float* cb3 = (const float*)d_in[13];

  const int N = in_sizes[0] / 3;  // 2048 rays

  char* p = (char*)d_ws;
  auto carve = [&](size_t bytes) -> void* {
    void* q = (void*)p;
    p += (bytes + 255) & ~(size_t)255;
    return q;
  };
  half_t* W2dp = (half_t*)carve(HDIM * HDIM * sizeof(half_t));
  half_t* W3dp = (half_t*)carve(HDIM * 16 * sizeof(half_t));
  half_t* W2cp = (half_t*)carve(HDIM * HDIM * sizeof(half_t));
  half_t* W3cp = (half_t*)carve(HDIM * 16 * sizeof(half_t));
  float*  zc   = (float*)carve((size_t)N * NS * sizeof(float));
  float*  zf   = (float*)carve((size_t)N * US * sizeof(float));
  float*  xyzc = (float*)carve((size_t)N * NS * 3 * sizeof(float));
  float*  xyzf = (float*)carve((size_t)N * US * 3 * sizeof(float));
  float*  sigc = (float*)carve((size_t)N * NS * sizeof(float));
  float*  sigf = (float*)carve((size_t)N * US * sizeof(float));
  half_t* geoc = (half_t*)carve((size_t)N * NS * GEO * sizeof(half_t));
  half_t* geof = (half_t*)carve((size_t)N * US * GEO * sizeof(half_t));
  int*    ordr = (int*)carve((size_t)N * TT * sizeof(int));
  float*  zsrt = (float*)carve((size_t)N * TT * sizeof(float));
  float*  wts  = (float*)carve((size_t)N * TT * sizeof(float));
  float*  rgbb = (float*)carve((size_t)N * TT * 3 * sizeof(float));

  // 1) pack weights into fragment-native f16 layout
  pack_b_kernel<<<(HDIM * HDIM + 255) / 256, 256, 0, stream>>>(dW2, W2dp, 16, HDIM, HDIM * HDIM);
  pack_b_kernel<<<(HDIM * 16 + 255) / 256, 256, 0, stream>>>(dW3, W3dp, 1, 16, HDIM * 16);
  pack_b_kernel<<<(HDIM * HDIM + 255) / 256, 256, 0, stream>>>(cW2, W2cp, 16, HDIM, HDIM * HDIM);
  pack_b_kernel<<<(HDIM * 16 + 255) / 256, 256, 0, stream>>>(cW3, W3cp, 1, 3, HDIM * 16);

  // 2) coarse samples
  ray_setup_kernel<<<(N * NS + 255) / 256, 256, 0, stream>>>(rays_o, rays_d, zc, xyzc, N * NS);

  // 3) coarse density MLP (WMMA)
  density_mlp_kernel<<<(N * NS) / (32 * PTILES), 512, 0, stream>>>(
      xyzc, dW1, db1, W2dp, db2, W3dp, db3, sigc, geoc);

  // 4) importance resample
  upsample_kernel<<<(N + 255) / 256, 256, 0, stream>>>(rays_o, rays_d, zc, sigc, zf, xyzf, N);

  // 5) fine density MLP (WMMA)
  density_mlp_kernel<<<(N * US) / (32 * PTILES), 512, 0, stream>>>(
      xyzf, dW1, db1, W2dp, db2, W3dp, db3, sigf, geof);

  // 6) stable merge + weights
  merge_kernel<<<(N + 255) / 256, 256, 0, stream>>>(zc, zf, sigc, sigf, zsrt, ordr, wts, N);

  // 7) color MLP (WMMA)
  color_mlp_kernel<<<(N * TT) / (32 * PTILES), 512, 0, stream>>>(
      rays_d, ordr, geoc, geof, cW1, cb1, W2cp, cb2, W3cp, cb3, rgbb);

  // 8) composite + distortion loss -> [N,5]
  composite_kernel<<<N, 256, 0, stream>>>(zsrt, wts, rgbb, (float*)d_out);
}
